// GeoGCNLayer_73212012528275
// MI455X (gfx1250) — matrix operand order repaired and checked by reading, not compiled, not run on hardware
//
#include <hip/hip_runtime.h>
#include <hip/hip_bf16.h>
#include <math.h>

typedef __attribute__((ext_vector_type(16))) _Float16 v16h;
typedef __attribute__((ext_vector_type(8)))  float    v8f;

#define DFEAT 64
#define HDIM  256

// Branch-free tanh: copysign(1 - 2/(exp(2|x|)+1), x).
// Saturates to +-1 for large |x| (exp -> inf, 2/inf -> 0). No EXEC branching.
__device__ __forceinline__ float fast_tanh(float x) {
    float ax = fabsf(x);
    float t  = __expf(ax + ax);
    float r  = 1.0f - 2.0f * __builtin_amdgcn_rcpf(t + 1.0f);
    return copysignf(r, x);
}

// ---------------------------------------------------------------- zero ws
__global__ void zero_kernel(float* __restrict__ p, size_t n) {
    size_t i = blockIdx.x * (size_t)blockDim.x + threadIdx.x;
    size_t stride = (size_t)gridDim.x * blockDim.x;
    for (; i < n; i += stride) p[i] = 0.0f;
}

// ---------------------------------------------------------------- scatter
// 16 threads per edge; float4 gather + 4 atomic f32 adds per thread.
// All accumulators are L2-resident on MI455X (192MB L2), so the 128M RMWs
// resolve at L2 throughput; no sorting/segmentation needed.
__global__ void scatter_kernel(const float* __restrict__ feat,
                               const int*   __restrict__ src,
                               const int*   __restrict__ dst,
                               const float* __restrict__ w,     // nullable
                               float*       __restrict__ acc,
                               float*       __restrict__ cnt,   // nullable
                               int E) {
    int t = blockIdx.x * blockDim.x + threadIdx.x;
    int e = t >> 4;
    if (e >= E) return;
    int lane = t & 15;
    int s = src[e];
    int d = dst[e];
    const float4* fr = (const float4*)(feat + (size_t)s * DFEAT);
    float4 v = fr[lane];
    if (w) {
        float wt = w[e];
        v.x *= wt; v.y *= wt; v.z *= wt; v.w *= wt;
    }
    float* out = acc + (size_t)d * DFEAT + lane * 4;
    atomicAdd(out + 0, v.x);
    atomicAdd(out + 1, v.y);
    atomicAdd(out + 2, v.z);
    atomicAdd(out + 3, v.w);
    if (cnt && lane == 0) atomicAdd(cnt + d, 1.0f);
}

// ---------------------------------------------------------------- geo mean
__global__ void finalize_geo(float* __restrict__ geo,
                             const float* __restrict__ cnt, int ND) {
    int idx = blockIdx.x * blockDim.x + threadIdx.x;
    if (idx >= ND) return;
    float c = cnt[idx >> 6];
    geo[idx] = (c > 0.0f) ? geo[idx] / c : 0.0f;
}

// ---------------------------------------------------------------- attention GEMM (WMMA)
// One wave handles a 16-row strip of one view (geo or trans):
//   h = tanh(z[16,64] @ W1[64,256] + b1);  s = h @ W2;  rs = sum(s)
// V_WMMA_F32_16X16X32_F16: 2 K-steps x 16 N-tiles = 32 WMMAs per wave.
//
// W1 is staged ONCE per block into LDS as f16, pre-swizzled into the exact
// B-fragment layout, so each inner iteration reads 2 fragments as contiguous
// 32B LDS vectors instead of 32 scattered global loads + converts.
//
// Fragment layouts (ISA 7.12.2):
//   A (16-bit 16x32): lane%16 = M; elem e -> K = (e<8?0:16) + (lane/16)*8 + (e&7)
//     derived: VGPR v=e/2, slot s=e&1, K = (v<4?0:16) + half*8 + 2*(v&3) + s
//   B (16-bit 32x16): lane%16 = N; elem e -> K = (e<8?0:16) + (lane/16)*8 + (e&7)
//   D (f32 16x16):    lane%16 = N; elem r -> M = r + 8*(lane/16)
__global__ void attn_kernel(const float* __restrict__ geo,
                            const float* __restrict__ trans,
                            const float* __restrict__ W1,
                            const float* __restrict__ b1v,
                            const float* __restrict__ W2,
                            float* __restrict__ wacc,
                            int Nn, int tiles) {
    // ---- cooperative W1 -> LDS f16 pre-swizzle (32 KB), before any exits
    __shared__ _Float16 w1s[DFEAT * HDIM];   // indexed as fragments
    for (int i = threadIdx.x; i < DFEAT * HDIM; i += blockDim.x) {
        int k    = i >> 8;          // 0..63
        int n    = i & 255;         // 0..255
        int j    = n >> 4;          // N-tile
        int ncol = n & 15;
        int kk   = k >> 5;          // K-step (0/1)
        int krem = k & 31;
        int half = (krem >> 3) & 1;
        int e    = (krem & 7) + ((krem >> 4) << 3);
        int lane = half * 16 + ncol;
        // fragment (j, kk), lane, element e -> contiguous 16-f16 run per lane
        w1s[((((j << 1) + kk) * 32) + lane) * 16 + e] = (_Float16)W1[i];
    }
    __syncthreads();

    int gt   = blockIdx.x * blockDim.x + threadIdx.x;
    int wid  = gt >> 5;             // wave-uniform (blockDim multiple of 32)
    int lane = gt & 31;
    if (wid >= 2 * tiles) return;   // whole-wave exit -> EXEC all-1s for WMMA

    int view = (wid >= tiles) ? 1 : 0;
    int tile = view ? (wid - tiles) : wid;
    const float* __restrict__ Z = view ? trans : geo;

    int half = lane >> 4;
    int m    = lane & 15;
    int arow = tile * 16 + m;

    // Load + cvt the two A fragments (K=0..31 and K=32..63) for this lane.
    v16h a0 = {}, a1 = {};
    if (arow < Nn) {
        const float* zr = Z + (size_t)arow * DFEAT;
#pragma unroll
        for (int e = 0; e < 16; ++e) {
            int v = e >> 1, s = e & 1;
            int k = ((v < 4) ? 0 : 16) + half * 8 + ((v & 3) << 1) + s;
            a0[e] = (_Float16)zr[k];
            a1[e] = (_Float16)zr[32 + k];
        }
    }

    const v16h* __restrict__ w1v = (const v16h*)w1s;   // 32B-aligned fragments

    float rs    = 0.0f;
    int   ncol  = lane & 15;
    int   mbase = half * 8;

    for (int j = 0; j < 16; ++j) {          // 16 N-tiles covering H=256
        int n = j * 16 + ncol;
        v16h bf0 = w1v[(j << 1) * 32 + lane];
        v16h bf1 = w1v[((j << 1) + 1) * 32 + lane];

        v8f c = {0.f, 0.f, 0.f, 0.f, 0.f, 0.f, 0.f, 0.f};
        c = __builtin_amdgcn_wmma_f32_16x16x32_f16(false, a0, false, bf0,
                                                   (short)0, c, false, false);
        c = __builtin_amdgcn_wmma_f32_16x16x32_f16(false, a1, false, bf1,
                                                   (short)0, c, false, false);
        float bb = b1v[n];
        float w2 = W2[n];
#pragma unroll
        for (int r = 0; r < 8; ++r) {
            int   rowg = tile * 16 + mbase + r;
            float val  = fast_tanh(c[r] + bb) * w2;
            rs += (rowg < Nn) ? val : 0.0f;   // cndmask, no EXEC branch
        }
    }

    // Wave32 butterfly reduction, then one atomic per wave.
#pragma unroll
    for (int off = 16; off > 0; off >>= 1)
        rs += __shfl_xor(rs, off, 32);
    if (lane == 0) atomicAdd(&wacc[view], rs);
}

// ---------------------------------------------------------------- beta blend
__global__ void out_kernel(const float* __restrict__ geo,
                           const float* __restrict__ trans,
                           const float* __restrict__ wacc,
                           float* __restrict__ out, int ND, int Nn) {
    int idx = blockIdx.x * blockDim.x + threadIdx.x;
    if (idx >= ND) return;
    float invN = 1.0f / (float)Nn;
    float w0 = wacc[0] * invN;
    float w1 = wacc[1] * invN;
    float mx = fmaxf(w0, w1);
    float e0 = __expf(w0 - mx);
    float e1 = __expf(w1 - mx);
    float inv = __builtin_amdgcn_rcpf(e0 + e1);
    out[idx] = (e0 * inv) * geo[idx] + (e1 * inv) * trans[idx];
}

// ---------------------------------------------------------------- launch
extern "C" void kernel_launch(void* const* d_in, const int* in_sizes, int n_in,
                              void* d_out, int out_size, void* d_ws, size_t ws_size,
                              hipStream_t stream) {
    const float* loc  = (const float*)d_in[0];
    const int*   gsrc = (const int*)  d_in[1];
    const int*   gdst = (const int*)  d_in[2];
    const int*   tsrc = (const int*)  d_in[3];
    const int*   tdst = (const int*)  d_in[4];
    const float* tw   = (const float*)d_in[5];
    const float* W1   = (const float*)d_in[6];
    const float* b1   = (const float*)d_in[7];
    const float* W2   = (const float*)d_in[8];

    const int ND = in_sizes[0];      // N * 64
    const int N  = ND / DFEAT;
    const int E  = in_sizes[1];

    float* ws    = (float*)d_ws;
    float* geo   = ws;                       // [N*64]
    float* trans = ws + (size_t)ND;          // [N*64]
    float* cnt   = ws + (size_t)2 * ND;      // [N]
    float* wacc  = cnt + N;                  // [2]

    // 1) zero accumulators (geo, trans, cnt, wacc)
    size_t zn = (size_t)2 * ND + N + 2;
    zero_kernel<<<2048, 256, 0, stream>>>(ws, zn);

    // 2) scatter phases (L2-resident atomics)
    int sblocks = (E * 16 + 255) / 256;
    scatter_kernel<<<sblocks, 256, 0, stream>>>(loc, gsrc, gdst, nullptr, geo, cnt, E);
    scatter_kernel<<<sblocks, 256, 0, stream>>>(loc, tsrc, tdst, tw, trans, nullptr, E);

    // 3) segment mean for geo
    finalize_geo<<<(ND + 255) / 256, 256, 0, stream>>>(geo, cnt, ND);

    // 4) fused semantic-attention logits via f16 WMMA + LDS-staged W1
    int tiles  = (N + 15) / 16;
    int waves  = 2 * tiles;
    int ablk   = (waves * 32 + 255) / 256;
    attn_kernel<<<ablk, 256, 0, stream>>>(geo, trans, W1, b1, W2, wacc, N, tiles);

    // 5) softmax over 2 logits + blend
    out_kernel<<<(ND + 255) / 256, 256, 0, stream>>>(geo, trans, wacc,
                                                     (float*)d_out, ND, N);
}
